// EncoderLayer_85718957294137
// MI455X (gfx1250) — compile-verified, matrix operand order checked by reading, not run on hardware
//
#include <hip/hip_runtime.h>

typedef __attribute__((ext_vector_type(16))) _Float16 v16h;
typedef __attribute__((ext_vector_type(8)))  float    v8f;

#define Bc   16
#define Lc   1536
#define Dc   512
#define DFFc 2048
#define Mc   (Bc * Lc)          // 24576 rows
#define TOPK 7
#define TT   8                  // tau tile for correlation kernel
#define BLD  ((size_t)Bc * Lc * Dc)
#define LDSP 40                 // padded row length (halves): 80B rows -> 16B-aligned,
                                // bank offset 20/row => conflict-free b128 LDS ops

// ---------------------------------------------------------------------------
// casts
// ---------------------------------------------------------------------------
__global__ void cast_f2h(const float* __restrict__ src, _Float16* __restrict__ dst, size_t n) {
  size_t i = (size_t)blockIdx.x * blockDim.x + threadIdx.x;
  if (i < n) dst[i] = (_Float16)src[i];
}

// W is [O][I][3]; extract tap t into dst[O][I]
__global__ void cast_w_stride3(const float* __restrict__ src, _Float16* __restrict__ dst,
                               int n, int t) {
  int i = blockIdx.x * blockDim.x + threadIdx.x;
  if (i < n) dst[i] = (_Float16)src[(size_t)i * 3 + t];
}

// ---------------------------------------------------------------------------
// WMMA GEMM:  out[m,n] = act( sum_k A[rowmap(m),k] * Bw[n,k] + bias[n] + resid[m,n] )
// A rows are circularly shifted by `shift` within each batch's Lc rows
// (turns the circular-pad conv1d into 3 accumulated GEMMs).
// Tile: 128(M) x 64(N), K-step 32. 256 threads = 8 wave32, wave grid 4x2,
// each wave computes a 32x32 patch = 2x2 v_wmma_f32_16x16x32_f16 tiles.
// Tiles staged with gfx1250 async memory->LDS copies (ASYNCcnt) in a
// software pipeline: stage k+1 is in flight while stage k's WMMAs run;
// the s_wait_asynccnt is placed AFTER the compute in program order.
// ---------------------------------------------------------------------------
__global__ __launch_bounds__(256)
void gemm_wmma(const _Float16* __restrict__ A, const _Float16* __restrict__ Bw,
               const float* __restrict__ bias, const float* __restrict__ resid,
               float* __restrict__ outF, _Float16* __restrict__ outH,
               int M, int N, int K, int Lrows, int shift, int relu) {
  __shared__ _Float16 As[2][128][LDSP];   // 20.5 KB
  __shared__ _Float16 Bs[2][64][LDSP];    // 10.25 KB

  const int tid  = threadIdx.x;
  const int lane = tid & 31;
  const int wave = tid >> 5;
  const int wm   = wave >> 1;        // 0..3
  const int wn   = wave & 1;         // 0..1
  const int tileM = blockIdx.x * 128;
  const int tileN = blockIdx.y * 64;
  const int lr = lane & 15;
  const int kb = (lane < 16) ? 0 : 8;   // per-lane K sub-offset (ISA A/B frag layout)

  // Per-thread async-copy assignments: 2 A chunks + 1 B chunk of 16B each.
  const int achunk0 = tid * 2;
  const int ar0 = achunk0 >> 2;
  const int ac0 = (achunk0 & 3) << 3;
  const int ar1 = (achunk0 + 1) >> 2;
  const int ac1 = ((achunk0 + 1) & 3) << 3;
  const int br  = tid >> 2;
  const int bcc = (tid & 3) << 3;

  // Pre-resolved circularly shifted global rows for the two A chunks.
  size_t arow0, arow1;
  {
    int m = tileM + ar0;
    int b = m / Lrows;
    int l = m - b * Lrows + shift;
    l += (l < 0) ? Lrows : 0;
    l -= (l >= Lrows) ? Lrows : 0;
    arow0 = (size_t)(b * Lrows + l);
    m = tileM + ar1;
    b = m / Lrows;
    l = m - b * Lrows + shift;
    l += (l < 0) ? Lrows : 0;
    l -= (l >= Lrows) ? Lrows : 0;
    arow1 = (size_t)(b * Lrows + l);
  }

  v8f c[2][2] = {};

  auto issue_stage = [&](int p, int k0) {
    unsigned long long ga0 = (unsigned long long)(A + (arow0 * K + k0 + ac0));
    unsigned lo0 = (unsigned)(size_t)(void*)&As[p][ar0][ac0];
    asm volatile("global_load_async_to_lds_b128 %0, %1, off"
                 :: "v"(lo0), "v"(ga0) : "memory");
    unsigned long long ga1 = (unsigned long long)(A + (arow1 * K + k0 + ac1));
    unsigned lo1 = (unsigned)(size_t)(void*)&As[p][ar1][ac1];
    asm volatile("global_load_async_to_lds_b128 %0, %1, off"
                 :: "v"(lo1), "v"(ga1) : "memory");
    unsigned long long gb = (unsigned long long)(Bw + ((size_t)(tileN + br) * K + k0 + bcc));
    unsigned lob = (unsigned)(size_t)(void*)&Bs[p][br][bcc];
    asm volatile("global_load_async_to_lds_b128 %0, %1, off"
                 :: "v"(lob), "v"(gb) : "memory");
  };

  union Frag { v16h v; float4 f[2]; };
  struct Frags { Frag a[2]; Frag b[2]; };

  auto load_frags = [&](int p, Frags& fr) {
#pragma unroll
    for (int i = 0; i < 2; ++i) {
      int row = wm * 32 + i * 16 + lr;
      fr.a[i].f[0] = *(const float4*)(&As[p][row][kb]);
      fr.a[i].f[1] = *(const float4*)(&As[p][row][kb + 16]);
    }
#pragma unroll
    for (int j = 0; j < 2; ++j) {
      int row = wn * 32 + j * 16 + lr;
      fr.b[j].f[0] = *(const float4*)(&Bs[p][row][kb]);
      fr.b[j].f[1] = *(const float4*)(&Bs[p][row][kb + 16]);
    }
  };

  const int nk = K >> 5;

  // ---- software-pipeline prologue ----
  issue_stage(0, 0);
  asm volatile("s_wait_asynccnt 0x0" ::: "memory");
  __syncthreads();
  Frags cur, nxt;
  load_frags(0, cur);
  __syncthreads();
  if (nk > 1) issue_stage(1, 32);     // stage 1 flies during stage-0 WMMAs

  for (int ks = 0; ks < nk; ++ks) {
    // compute stage ks (async copies for stage ks+1 are in flight)
#pragma unroll
    for (int i = 0; i < 2; ++i)
#pragma unroll
      for (int j = 0; j < 2; ++j)
        c[i][j] = __builtin_amdgcn_wmma_f32_16x16x32_f16(
            false, cur.a[i].v, false, cur.b[j].v, (short)0, c[i][j], false, false);

    if (ks + 1 < nk) {
      asm volatile("s_wait_asynccnt 0x0" ::: "memory");  // stage ks+1 landed
      __syncthreads();   // barrier a: all waves' stage-(ks+1) data visible;
                         // reachable only after each wave's dscnt-wait for its
                         // buffer-p fragment reads (forced by the WMMAs above)
      load_frags((ks + 1) & 1, nxt);
      __syncthreads();   // barrier b: all reads of buffer (ks&1) retired
      if (ks + 2 < nk) issue_stage(ks & 1, (ks + 2) << 5);
      cur = nxt;
    }
  }

  // Epilogue: C/D layout — VGPR r: lanes 0-15 hold M=r, lanes 16-31 hold M=8+r; N=lane&15
#pragma unroll
  for (int i = 0; i < 2; ++i) {
    int mBase = tileM + wm * 32 + i * 16 + ((lane < 16) ? 0 : 8);
#pragma unroll
    for (int j = 0; j < 2; ++j) {
      int n = tileN + wn * 32 + j * 16 + lr;
      float bb = bias ? bias[n] : 0.0f;
#pragma unroll
      for (int r = 0; r < 8; ++r) {
        int m = mBase + r;
        float val = c[i][j][r] + bb;
        if (resid) val += resid[(size_t)m * N + n];
        if (relu && val < 0.0f) val = 0.0f;
        if (outF) outF[(size_t)m * N + n] = val;
        if (outH) outH[(size_t)m * N + n] = (_Float16)val;
      }
    }
  }
}

// ---------------------------------------------------------------------------
// mean autocorrelation:  mc[b,tau] = sum_l sum_c Q[b,(l+tau)%L,c] * K[b,l,c]
// (the 1/D mean-over-channels factor is applied in the softmax kernel)
// ---------------------------------------------------------------------------
__global__ __launch_bounds__(256)
void corr_kernel(const float* __restrict__ Q, const float* __restrict__ Kx,
                 float* __restrict__ mc) {
  __shared__ float red[256];
  const int b   = blockIdx.y;
  const int t0  = blockIdx.x * TT;
  const int tid = threadIdx.x;
  const float* Qb = Q  + (size_t)b * Lc * Dc;
  const float* Kb = Kx + (size_t)b * Lc * Dc;

  float acc[TT];
#pragma unroll
  for (int t = 0; t < TT; ++t) acc[t] = 0.0f;

  for (int l = tid; l < Lc; l += 256) {
    const float* kr = Kb + (size_t)l * Dc;
    const float* qr[TT];
#pragma unroll
    for (int t = 0; t < TT; ++t) {
      int lq = l + t0 + t;
      if (lq >= Lc) lq -= Lc;
      qr[t] = Qb + (size_t)lq * Dc;
    }
    for (int cix = 0; cix < Dc; cix += 4) {
      float4 kv = *(const float4*)(kr + cix);
#pragma unroll
      for (int t = 0; t < TT; ++t) {
        float4 qv = *(const float4*)(qr[t] + cix);
        acc[t] += qv.x * kv.x + qv.y * kv.y + qv.z * kv.z + qv.w * kv.w;
      }
    }
  }
  for (int t = 0; t < TT; ++t) {
    red[tid] = acc[t];
    __syncthreads();
    for (int s = 128; s > 0; s >>= 1) {
      if (tid < s) red[tid] += red[tid + s];
      __syncthreads();
    }
    if (tid == 0) mc[(size_t)b * Lc + t0 + t] = red[0];
    __syncthreads();
  }
}

// ---------------------------------------------------------------------------
// top-k (k=7) over batch-mean correlation + per-batch softmax of the selected
// delays (single workgroup).
// ---------------------------------------------------------------------------
__global__ __launch_bounds__(256)
void topk_softmax_kernel(const float* __restrict__ mc, float* __restrict__ wsoft,
                         int* __restrict__ idxout, float scale) {
  __shared__ float g[Lc];
  __shared__ float rv[256];
  __shared__ int   ri[256];
  __shared__ int   sidx[TOPK];
  const int tid = threadIdx.x;

  for (int t = tid; t < Lc; t += 256) {
    float s = 0.0f;
    for (int b = 0; b < Bc; ++b) s += mc[(size_t)b * Lc + t];
    g[t] = s;
  }
  __syncthreads();

  for (int i = 0; i < TOPK; ++i) {
    float bv = -3.0e38f; int bi = 0;
    for (int t = tid; t < Lc; t += 256)
      if (g[t] > bv) { bv = g[t]; bi = t; }
    rv[tid] = bv; ri[tid] = bi;
    __syncthreads();
    for (int s = 128; s > 0; s >>= 1) {
      if (tid < s && rv[tid + s] > rv[tid]) { rv[tid] = rv[tid + s]; ri[tid] = ri[tid + s]; }
      __syncthreads();
    }
    if (tid == 0) { sidx[i] = ri[0]; g[ri[0]] = -3.0e38f; }
    __syncthreads();
  }

  if (tid < TOPK) idxout[tid] = sidx[tid];
  if (tid < Bc) {
    float vals[TOPK];
    float mx = -3.0e38f;
    for (int i = 0; i < TOPK; ++i) {
      vals[i] = mc[(size_t)tid * Lc + sidx[i]] * scale;
      if (vals[i] > mx) mx = vals[i];
    }
    float se = 0.0f;
    for (int i = 0; i < TOPK; ++i) { vals[i] = __expf(vals[i] - mx); se += vals[i]; }
    for (int i = 0; i < TOPK; ++i) wsoft[tid * TOPK + i] = vals[i] / se;
  }
}

// ---------------------------------------------------------------------------
// time-delay aggregation: out[b,l,c] = sum_i w[b,i] * V[b,(l+idx[i])%L,c]
// ---------------------------------------------------------------------------
__global__ void agg_kernel(const float* __restrict__ V, const float* __restrict__ wsoft,
                           const int* __restrict__ idx, float* __restrict__ out) {
  size_t i4 = (size_t)blockIdx.x * blockDim.x + threadIdx.x;
  size_t total = BLD / 4;
  if (i4 >= total) return;
  size_t i   = i4 * 4;
  int    b   = (int)(i / ((size_t)Lc * Dc));
  size_t rem = i - (size_t)b * Lc * Dc;
  int    l   = (int)(rem / Dc);
  int    cix = (int)(rem - (size_t)l * Dc);
  const float* Vb = V + (size_t)b * Lc * Dc;
  float ax = 0.f, ay = 0.f, az = 0.f, aw = 0.f;
#pragma unroll
  for (int t = 0; t < TOPK; ++t) {
    int d  = idx[t];
    int ls = l + d;
    if (ls >= Lc) ls -= Lc;
    float w = wsoft[b * TOPK + t];
    float4 vv = *(const float4*)(Vb + (size_t)ls * Dc + cix);
    ax += w * vv.x; ay += w * vv.y; az += w * vv.z; aw += w * vv.w;
  }
  float4 o; o.x = ax; o.y = ay; o.z = az; o.w = aw;
  *(float4*)(out + i) = o;
}

// ---------------------------------------------------------------------------
// series_decomp: dst = x - moving_avg_25(x) (replicate padding), running window
// one thread per (batch, channel)
// ---------------------------------------------------------------------------
__global__ void decomp_kernel(const float* __restrict__ x, float* __restrict__ dst) {
  int t = blockIdx.x * blockDim.x + threadIdx.x;
  if (t >= Bc * Dc) return;
  int b   = t / Dc;
  int cix = t - b * Dc;
  const float* xb = x   + (size_t)b * Lc * Dc + cix;
  float*       db = dst + (size_t)b * Lc * Dc + cix;
  float sum = 0.0f;
  for (int j = -12; j <= 12; ++j) {
    int jj = j < 0 ? 0 : j;
    sum += xb[(size_t)jj * Dc];
  }
  for (int l = 0; l < Lc; ++l) {
    db[(size_t)l * Dc] = xb[(size_t)l * Dc] - sum * (1.0f / 25.0f);
    int add = l + 13; if (add > Lc - 1) add = Lc - 1;
    int rem = l - 12; if (rem < 0) rem = 0;
    sum += xb[(size_t)add * Dc] - xb[(size_t)rem * Dc];
  }
}

// ---------------------------------------------------------------------------
// host driver
// ---------------------------------------------------------------------------
extern "C" void kernel_launch(void* const* d_in, const int* in_sizes, int n_in,
                              void* d_out, int out_size, void* d_ws, size_t ws_size,
                              hipStream_t stream) {
  (void)in_sizes; (void)n_in; (void)out_size; (void)ws_size;

  const float* x_s_in   = (const float*)d_in[0];
  const float* x_w_in   = (const float*)d_in[1];
  const float* wc1_W    = (const float*)d_in[2];
  const float* wc1_b    = (const float*)d_in[3];
  const float* wc2_W    = (const float*)d_in[4];
  const float* wc2_b    = (const float*)d_in[5];
  const float* attn_Wm  = (const float*)d_in[6];
  const float* attn_bm  = (const float*)d_in[7];
  const float* wc1_proj = (const float*)d_in[8];
  const float* wc2_proj = (const float*)d_in[9];
  const float* conv1_W  = (const float*)d_in[10];
  const float* conv2_W  = (const float*)d_in[11];
  float* out_res = (float*)d_out;
  float* out_xw  = (float*)d_out + BLD;

  char* wp = (char*)d_ws;
  auto alloc = [&](size_t bytes) -> char* {
    char* p = wp; wp += (bytes + 255) & ~(size_t)255; return p;
  };
  float*    xs    = (float*)alloc(BLD * 4);
  float*    qb    = (float*)alloc(BLD * 4);
  float*    kb    = (float*)alloc(BLD * 4);
  float*    vb    = (float*)alloc(BLD * 4);
  float*    t0    = (float*)alloc(BLD * 4);
  float*    t1    = (float*)alloc(BLD * 4);
  float*    xw1   = (float*)alloc(BLD * 4);
  _Float16* a16   = (_Float16*)alloc((size_t)Mc * Dc * 2);
  _Float16* b16   = (_Float16*)alloc((size_t)Mc * DFFc * 2);
  _Float16* w16   = (_Float16*)alloc((size_t)DFFc * Dc * 2);
  float*    mcb   = (float*)alloc((size_t)Bc * Lc * 4);
  float*    wsoft = (float*)alloc((size_t)Bc * TOPK * 4);
  int*      idxb  = (int*)alloc(64);

  auto cast = [&](const float* s, _Float16* d, size_t n) {
    int blocks = (int)((n + 255) / 256);
    cast_f2h<<<dim3(blocks), dim3(256), 0, stream>>>(s, d, n);
  };
  auto gemm = [&](const _Float16* A, const _Float16* Bw, const float* bias,
                  const float* resid, float* oF, _Float16* oH, int N, int K,
                  int shift, int relu) {
    dim3 grid(Mc / 128, N / 64);
    gemm_wmma<<<grid, dim3(256), 0, stream>>>(A, Bw, bias, resid, oF, oH,
                                              Mc, N, K, Lc, shift, relu);
  };
  // AutoCorrelation layer; head count is mathematically irrelevant because
  // corr is averaged over all H*E = D channels and aggregation is per-channel.
  auto attn = [&](const float* qin, const float* kvin, const float* W, const float* bias,
                  const float* residual, float* xs_out) {
    cast(qin, a16, BLD);
    cast(W + 0 * (size_t)Dc * Dc, w16, (size_t)Dc * Dc);
    gemm(a16, w16, bias + 0 * Dc, nullptr, qb, nullptr, Dc, Dc, 0, 0);
    cast(kvin, a16, BLD);
    cast(W + 1 * (size_t)Dc * Dc, w16, (size_t)Dc * Dc);
    gemm(a16, w16, bias + 1 * Dc, nullptr, kb, nullptr, Dc, Dc, 0, 0);
    cast(W + 2 * (size_t)Dc * Dc, w16, (size_t)Dc * Dc);
    gemm(a16, w16, bias + 2 * Dc, nullptr, vb, nullptr, Dc, Dc, 0, 0);

    corr_kernel<<<dim3(Lc / TT, Bc), dim3(256), 0, stream>>>(qb, kb, mcb);
    topk_softmax_kernel<<<dim3(1), dim3(256), 0, stream>>>(mcb, wsoft, idxb,
                                                           1.0f / (float)Dc);
    {
      int blocks = (int)((BLD / 4 + 255) / 256);
      agg_kernel<<<dim3(blocks), dim3(256), 0, stream>>>(vb, wsoft, idxb, t0);
    }
    cast(t0, a16, BLD);
    cast(W + 3 * (size_t)Dc * Dc, w16, (size_t)Dc * Dc);
    gemm(a16, w16, bias + 3 * Dc, residual, xs_out, nullptr, Dc, Dc, 0, 0);
  };
  // circular conv1d k=3 == 3 accumulated GEMMs with row shift t-1
  auto conv = [&](const float* xin, const float* Wc, float* outp) {
    cast(xin, a16, BLD);
    for (int t = 0; t < 3; ++t) {
      int blocks = (Dc * Dc + 255) / 256;
      cast_w_stride3<<<dim3(blocks), dim3(256), 0, stream>>>(Wc, w16, Dc * Dc, t);
      gemm(a16, w16, nullptr, (t ? outp : nullptr), outp, nullptr, Dc, Dc, t - 1, 0);
    }
  };

  // -------- pipeline (matches reference order) --------
  attn(x_w_in, x_s_in, wc1_W, wc1_b, x_s_in, xs);              // x_s += wc_decomp1
  conv(x_w_in, wc1_proj, xw1);                                 // x_w = token_embed1
  attn(xs, xs, attn_Wm, attn_bm, xs, xs);                      // x_s += self-attn
  decomp_kernel<<<dim3((Bc * Dc + 255) / 256), dim3(256), 0, stream>>>(xs, t1);
  attn(xw1, t1, wc2_W, wc2_b, t1, xs);                         // x_s += wc_decomp2
  conv(xw1, wc2_proj, out_xw);                                 // x_w_new
  // FFN: hidden stored directly as f16 (relu fused), output fuses residual
  cast(xs, a16, BLD);
  cast(conv1_W, w16, (size_t)DFFc * Dc);
  gemm(a16, w16, nullptr, nullptr, nullptr, b16, DFFc, Dc, 0, 1);
  cast(conv2_W, w16, (size_t)Dc * DFFc);
  gemm(b16, w16, nullptr, xs, t0, nullptr, Dc, DFFc, 0, 0);
  decomp_kernel<<<dim3((Bc * Dc + 255) / 256), dim3(256), 0, stream>>>(t0, out_res);
}
